// GraphTransformerBaseBlock_91113436217869
// MI455X (gfx1250) — matrix operand order repaired and checked by reading, not compiled, not run on hardware
//
#include <hip/hip_runtime.h>
#include <hip/hip_bf16.h>
#include <math.h>

// ---------------- types ----------------
typedef __bf16 bf16;
typedef __attribute__((ext_vector_type(16))) __bf16 v16bf;
typedef __attribute__((ext_vector_type(8)))  __bf16 v8bf;
typedef __attribute__((ext_vector_type(8)))  float  v8f;

#define C_DIM 128
#define H_HEADS 8
#define D_HEAD 16
#define E_DIM 64
#define HID_DIM 512

// NOTE: tile scheduling assumes N % 16 == 0 and E % 64 == 0 (true here: 50000, 800000).

// ---------------- helpers ----------------
__device__ __forceinline__ bf16 f2bf(float f) {
  unsigned u = __float_as_uint(f);
  unsigned r = (u + 0x7FFFu + ((u >> 16) & 1u)) >> 16;
  unsigned short s = (unsigned short)r;
  return __builtin_bit_cast(bf16, s);
}

// monotone float<->uint key for atomicMax-based segment max
__device__ __forceinline__ unsigned fkey(float f) {
  unsigned u = __float_as_uint(f);
  return (u & 0x80000000u) ? ~u : (u | 0x80000000u);
}
__device__ __forceinline__ float fkey_dec(unsigned k) {
  unsigned u = (k & 0x80000000u) ? (k & 0x7FFFFFFFu) : ~k;
  return __uint_as_float(u);
}

__device__ __forceinline__ float wave_sum(float v) {
#pragma unroll
  for (int m = 16; m >= 1; m >>= 1) v += __shfl_xor(v, m, 32);
  return v;
}

__device__ __forceinline__ float gelu_tanh(float x) {
  float x3 = x * x * x;
  return 0.5f * x * (1.0f + tanhf(0.7978845608028654f * (x + 0.044715f * x3)));
}

// A fragment (16x32 bf16) from an LDS tile, row-major, leading dim ldm.
// Lane l<16: row l, K = k0..k0+7 (lo) and k0+16..k0+23 (hi)
// Lane l>=16: row l-16, K = k0+8..k0+15 (lo) and k0+24..k0+31 (hi)
__device__ __forceinline__ v16bf load_a_frag(const bf16* s, int ldm, int k0, int lane) {
  int row = lane & 15;
  int kb = k0 + ((lane & 16) ? 8 : 0);
  const bf16* p = s + row * ldm + kb;
  v8bf lo = *(const v8bf*)p;
  v8bf hi = *(const v8bf*)(p + 16);
  v16bf a;
#pragma unroll
  for (int i = 0; i < 8; ++i) { a[i] = lo[i]; a[i + 8] = hi[i]; }
  return a;
}

// B fragment from pre-packed weights: fragment (ktile,ntile), 32 lanes * 16 bf16 contiguous.
__device__ __forceinline__ v16bf load_b_frag(const bf16* packed, int ktile, int ntile,
                                             int ntiles, int lane) {
  return *(const v16bf*)(packed + (((size_t)ktile * ntiles + ntile) * 32 + lane) * 16);
}

__device__ __forceinline__ v8f wmma_bf16(v16bf a, v16bf b, v8f c) {
  return __builtin_amdgcn_wmma_f32_16x16x32_bf16(false, a, false, b, (short)0, c, false, false);
}

// stage one 16x16 f32 accumulator tile (bias added) into a 16x128 LDS tile
__device__ __forceinline__ void stage_acc(float* sO, v8f acc, int nt, int lane, float bias) {
  int col = nt * 16 + (lane & 15);
  int ro = (lane >> 4) * 8;
#pragma unroll
  for (int r = 0; r < 8; ++r) sO[(r + ro) * C_DIM + col] = acc[r] + bias;
}

// wave-coalesced store of a 16x128 f32 tile (same-wave LDS ops are in-order)
__device__ __forceinline__ void wave_store_tile(const float* sO, float* dst, int lane) {
#pragma unroll
  for (int j = 0; j < 16; ++j) {
    float4 t = *(const float4*)(sO + j * C_DIM + lane * 4);
    *(float4*)(dst + (size_t)j * C_DIM + lane * 4) = t;
  }
}

// ---------------- K0: pack fp32 weight [K,N] -> bf16 WMMA-B fragment order ----------------
__global__ void pack_weight(const float* __restrict__ W, bf16* __restrict__ P, int K, int Nout) {
  int idx = blockIdx.x * blockDim.x + threadIdx.x;
  int ntiles = Nout >> 4, ktiles = K >> 5;
  int total = ktiles * ntiles * 32;
  if (idx >= total) return;
  int lane = idx & 31, tile = idx >> 5;
  int kt = tile / ntiles, nt = tile - kt * ntiles;
  int col = (nt << 4) + (lane & 15);
  int kr = (kt << 5) + ((lane & 16) ? 8 : 0);
  v16bf frag;
#pragma unroll
  for (int i = 0; i < 16; ++i) frag[i] = f2bf(W[(size_t)(kr + i + (i >= 8 ? 8 : 0)) * Nout + col]);
  *(v16bf*)(P + (size_t)idx * 16) = frag;
}

// ---------------- K1: LayerNorm1 + q/k/v/lin_self GEMMs (one 16-row tile / block) ----------
__global__ __launch_bounds__(128) void node_ln_qkvs(
    const float* __restrict__ x, const float* __restrict__ g, const float* __restrict__ bln,
    const bf16* __restrict__ pq, const bf16* __restrict__ pk,
    const bf16* __restrict__ pv, const bf16* __restrict__ ps,
    const float* __restrict__ bq, const float* __restrict__ bk,
    const float* __restrict__ bv, const float* __restrict__ bs,
    float* __restrict__ q, float* __restrict__ k, float* __restrict__ v,
    float* __restrict__ xr) {
  __shared__ bf16 sA[16 * C_DIM];
  __shared__ float sOut[4 * 16 * C_DIM];   // one 16x128 f32 tile per wave
  int row0 = blockIdx.x * 16;
  int wave = threadIdx.x >> 5, lane = threadIdx.x & 31;

  for (int r = wave * 4; r < wave * 4 + 4; ++r) {
    float4 xv = *(const float4*)(x + (size_t)(row0 + r) * C_DIM + lane * 4);
    float mu = wave_sum(xv.x + xv.y + xv.z + xv.w) * (1.0f / 128.0f);
    float d0 = xv.x - mu, d1 = xv.y - mu, d2 = xv.z - mu, d3 = xv.w - mu;
    float var = wave_sum(d0 * d0 + d1 * d1 + d2 * d2 + d3 * d3) * (1.0f / 128.0f);
    float rs = rsqrtf(var + 1e-5f);
    float4 gv = *(const float4*)(g + lane * 4);
    float4 bb = *(const float4*)(bln + lane * 4);
    bf16* sp = sA + r * C_DIM + lane * 4;
    sp[0] = f2bf(d0 * rs * gv.x + bb.x);
    sp[1] = f2bf(d1 * rs * gv.y + bb.y);
    sp[2] = f2bf(d2 * rs * gv.z + bb.z);
    sp[3] = f2bf(d3 * rs * gv.w + bb.w);
  }
  __syncthreads();

  const bf16* P; const float* B; float* O;
  if      (wave == 0) { P = pq; B = bq; O = q;  }
  else if (wave == 1) { P = pk; B = bk; O = k;  }
  else if (wave == 2) { P = pv; B = bv; O = v;  }
  else                { P = ps; B = bs; O = xr; }

  // hoist A fragments (reused for all 8 ntiles)
  v16bf a0 = load_a_frag(sA, C_DIM, 0,  lane);
  v16bf a1 = load_a_frag(sA, C_DIM, 32, lane);
  v16bf a2 = load_a_frag(sA, C_DIM, 64, lane);
  v16bf a3 = load_a_frag(sA, C_DIM, 96, lane);

  float* sO = sOut + wave * 16 * C_DIM;
#pragma unroll
  for (int nt = 0; nt < 8; ++nt) {
    v8f acc = {};
    acc = wmma_bf16(a0, load_b_frag(P, 0, nt, 8, lane), acc);
    acc = wmma_bf16(a1, load_b_frag(P, 1, nt, 8, lane), acc);
    acc = wmma_bf16(a2, load_b_frag(P, 2, nt, 8, lane), acc);
    acc = wmma_bf16(a3, load_b_frag(P, 3, nt, 8, lane), acc);
    stage_acc(sO, acc, nt, lane, B[nt * 16 + (lane & 15)]);
  }
  wave_store_tile(sO, O + (size_t)row0 * C_DIM, lane);
}

// ---------------- K2: e = edge_attr @ we + be   (64 edges / block) ----------------
__global__ __launch_bounds__(128) void edge_lin(
    const float* __restrict__ ea, const bf16* __restrict__ pe, const float* __restrict__ be,
    float* __restrict__ emat) {
  __shared__ bf16 sE[64 * E_DIM];
  __shared__ float sOut[4 * 16 * C_DIM];
  int e0 = blockIdx.x * 64;
  int wave = threadIdx.x >> 5, lane = threadIdx.x & 31;

#pragma unroll
  for (int j = 0; j < 8; ++j) {
    int flat = (j * 128 + threadIdx.x) * 4;     // coalesced 16B per thread
    float4 t = *(const float4*)(ea + (size_t)e0 * E_DIM + flat);
    sE[flat + 0] = f2bf(t.x); sE[flat + 1] = f2bf(t.y);
    sE[flat + 2] = f2bf(t.z); sE[flat + 3] = f2bf(t.w);
  }
  __syncthreads();

  const bf16* sRow = sE + wave * 16 * E_DIM;
  v16bf a0 = load_a_frag(sRow, E_DIM, 0,  lane);
  v16bf a1 = load_a_frag(sRow, E_DIM, 32, lane);

  float* sO = sOut + wave * 16 * C_DIM;
#pragma unroll
  for (int nt = 0; nt < 8; ++nt) {
    v8f acc = {};
    acc = wmma_bf16(a0, load_b_frag(pe, 0, nt, 8, lane), acc);
    acc = wmma_bf16(a1, load_b_frag(pe, 1, nt, 8, lane), acc);
    stage_acc(sO, acc, nt, lane, be[nt * 16 + (lane & 15)]);
  }
  wave_store_tile(sO, emat + (size_t)(e0 + wave * 16) * C_DIM, lane);
}

// ---------------- K3a: init segment buffers ----------------
__global__ void attn_init(unsigned* __restrict__ amax, float* __restrict__ den,
                          float* __restrict__ num, int Nn) {
  int idx = blockIdx.x * blockDim.x + threadIdx.x;
  if (idx < Nn * C_DIM) num[idx] = 0.0f;
  if (idx < Nn * H_HEADS) { den[idx] = 0.0f; amax[idx] = 0u; }
}

// ---------------- K3: alpha + segment max (one thread per edge-head) ----------------
__global__ __launch_bounds__(256) void attn_alpha(
    const float* __restrict__ q, const float* __restrict__ k, const float* __restrict__ emat,
    const long long* __restrict__ eidx, float* __restrict__ alpha,
    unsigned* __restrict__ amax, int Ee) {
  int idx = blockIdx.x * blockDim.x + threadIdx.x;
  if (idx >= Ee * H_HEADS) return;
  int e = idx >> 3, h = idx & 7;
  long long s = eidx[e], d = eidx[(size_t)Ee + e];
  const float* qp = q    + (size_t)d * C_DIM + h * D_HEAD;
  const float* kp = k    + (size_t)s * C_DIM + h * D_HEAD;
  const float* ep = emat + (size_t)e * C_DIM + h * D_HEAD;
  float acc = 0.f;
#pragma unroll
  for (int i = 0; i < 4; ++i) {
    float4 qv = *(const float4*)(qp + i * 4);
    float4 kv = *(const float4*)(kp + i * 4);
    float4 ev = *(const float4*)(ep + i * 4);
    acc += qv.x * (kv.x + ev.x) + qv.y * (kv.y + ev.y) +
           qv.z * (kv.z + ev.z) + qv.w * (kv.w + ev.w);
  }
  float a = acc * 0.25f;   // 1/sqrt(16)
  alpha[idx] = a;
  atomicMax(&amax[(size_t)d * H_HEADS + h], fkey(a));
}

// ---------------- K4: exp + segment sum of den and numerator ----------------
__global__ __launch_bounds__(256) void attn_accum(
    const float* __restrict__ alpha, const unsigned* __restrict__ amax,
    const float* __restrict__ v, const float* __restrict__ emat,
    const long long* __restrict__ eidx, float* __restrict__ den,
    float* __restrict__ num, int Ee) {
  int idx = blockIdx.x * blockDim.x + threadIdx.x;
  if (idx >= Ee * H_HEADS) return;
  int e = idx >> 3, h = idx & 7;
  long long s = eidx[e], d = eidx[(size_t)Ee + e];
  float m = fkey_dec(amax[(size_t)d * H_HEADS + h]);
  float ex = __expf(alpha[idx] - m);
  atomicAdd(&den[(size_t)d * H_HEADS + h], ex);
  const float* vp = v    + (size_t)s * C_DIM + h * D_HEAD;
  const float* ep = emat + (size_t)e * C_DIM + h * D_HEAD;
  float* np = num + (size_t)d * C_DIM + h * D_HEAD;
#pragma unroll
  for (int i = 0; i < D_HEAD; ++i) atomicAdd(&np[i], ex * (vp[i] + ep[i]));
}

// ---------------- K5: (num/den + x_r) @ wp + bp + x_skip ----------------
__global__ __launch_bounds__(128) void proj(
    const float* __restrict__ num, const float* __restrict__ den,
    const float* __restrict__ xr, const float* __restrict__ x,
    const bf16* __restrict__ pw, const float* __restrict__ bp,
    float* __restrict__ y) {
  __shared__ bf16 sA[16 * C_DIM];
  __shared__ float sO[16 * C_DIM];
  int row0 = blockIdx.x * 16;
  int wave = threadIdx.x >> 5, lane = threadIdx.x & 31;

  for (int r = wave * 4; r < wave * 4 + 4; ++r) {
    size_t row = row0 + r;
    int c = lane * 4;
    float4 nv = *(const float4*)(num + row * C_DIM + c);
    float4 rv = *(const float4*)(xr  + row * C_DIM + c);
    float inv = 1.0f / (den[row * H_HEADS + (c >> 4)] + 1e-16f);
    bf16* sp = sA + r * C_DIM + c;
    sp[0] = f2bf(nv.x * inv + rv.x); sp[1] = f2bf(nv.y * inv + rv.y);
    sp[2] = f2bf(nv.z * inv + rv.z); sp[3] = f2bf(nv.w * inv + rv.w);
  }
  __syncthreads();

  v16bf a0 = load_a_frag(sA, C_DIM, 0,  lane);
  v16bf a1 = load_a_frag(sA, C_DIM, 32, lane);
  v16bf a2 = load_a_frag(sA, C_DIM, 64, lane);
  v16bf a3 = load_a_frag(sA, C_DIM, 96, lane);

#pragma unroll
  for (int t = 0; t < 2; ++t) {
    int nt = wave * 2 + t;
    v8f acc = {};
    acc = wmma_bf16(a0, load_b_frag(pw, 0, nt, 8, lane), acc);
    acc = wmma_bf16(a1, load_b_frag(pw, 1, nt, 8, lane), acc);
    acc = wmma_bf16(a2, load_b_frag(pw, 2, nt, 8, lane), acc);
    acc = wmma_bf16(a3, load_b_frag(pw, 3, nt, 8, lane), acc);
    stage_acc(sO, acc, nt, lane, bp[nt * 16 + (lane & 15)]);
  }
  __syncthreads();

  // coalesced: y = proj + x_skip
  for (int r = wave * 4; r < wave * 4 + 4; ++r) {
    size_t row = row0 + r;
    int c = lane * 4;
    float4 pv4 = *(const float4*)(sO + r * C_DIM + c);
    float4 xv  = *(const float4*)(x + row * C_DIM + c);
    float4 o   = make_float4(pv4.x + xv.x, pv4.y + xv.y, pv4.z + xv.z, pv4.w + xv.w);
    *(float4*)(y + row * C_DIM + c) = o;
  }
}

// ---------------- K6: LN2 + MLP (128->512 gelu ->128) + residual ----------------
__global__ __launch_bounds__(128) void mlp(
    const float* __restrict__ y, const float* __restrict__ g2, const float* __restrict__ bl2,
    const bf16* __restrict__ p1, const float* __restrict__ b1,
    const bf16* __restrict__ p2, const float* __restrict__ b2,
    float* __restrict__ out) {
  __shared__ bf16 sA[16 * C_DIM];
  __shared__ bf16 sH[16 * HID_DIM];
  __shared__ float sO[16 * C_DIM];
  int row0 = blockIdx.x * 16;
  int wave = threadIdx.x >> 5, lane = threadIdx.x & 31;

  for (int r = wave * 4; r < wave * 4 + 4; ++r) {
    float4 xv = *(const float4*)(y + (size_t)(row0 + r) * C_DIM + lane * 4);
    float mu = wave_sum(xv.x + xv.y + xv.z + xv.w) * (1.0f / 128.0f);
    float d0 = xv.x - mu, d1 = xv.y - mu, d2 = xv.z - mu, d3 = xv.w - mu;
    float var = wave_sum(d0 * d0 + d1 * d1 + d2 * d2 + d3 * d3) * (1.0f / 128.0f);
    float rs = rsqrtf(var + 1e-5f);
    float4 gv = *(const float4*)(g2 + lane * 4);
    float4 bb = *(const float4*)(bl2 + lane * 4);
    bf16* sp = sA + r * C_DIM + lane * 4;
    sp[0] = f2bf(d0 * rs * gv.x + bb.x);
    sp[1] = f2bf(d1 * rs * gv.y + bb.y);
    sp[2] = f2bf(d2 * rs * gv.z + bb.z);
    sp[3] = f2bf(d3 * rs * gv.w + bb.w);
  }
  __syncthreads();

  int col = lane & 15, ro = (lane >> 4) * 8;

  // GEMM1: [16,128] @ [128,512], tanh-GELU, stash bf16 in LDS
  v16bf a0 = load_a_frag(sA, C_DIM, 0,  lane);
  v16bf a1 = load_a_frag(sA, C_DIM, 32, lane);
  v16bf a2 = load_a_frag(sA, C_DIM, 64, lane);
  v16bf a3 = load_a_frag(sA, C_DIM, 96, lane);
#pragma unroll
  for (int t = 0; t < 8; ++t) {
    int nt = wave * 8 + t;
    v8f acc = {};
    acc = wmma_bf16(a0, load_b_frag(p1, 0, nt, 32, lane), acc);
    acc = wmma_bf16(a1, load_b_frag(p1, 1, nt, 32, lane), acc);
    acc = wmma_bf16(a2, load_b_frag(p1, 2, nt, 32, lane), acc);
    acc = wmma_bf16(a3, load_b_frag(p1, 3, nt, 32, lane), acc);
    int c = nt * 16 + col;
    float bb = b1[c];
#pragma unroll
    for (int r = 0; r < 8; ++r)
      sH[(r + ro) * HID_DIM + c] = f2bf(gelu_tanh(acc[r] + bb));
  }
  __syncthreads();

  // GEMM2: [16,512] @ [512,128]
#pragma unroll
  for (int t = 0; t < 2; ++t) {
    int nt = wave * 2 + t;
    v8f acc = {};
#pragma unroll
    for (int kt = 0; kt < 16; ++kt)
      acc = wmma_bf16(load_a_frag(sH, HID_DIM, kt * 32, lane), load_b_frag(p2, kt, nt, 8, lane), acc);
    stage_acc(sO, acc, nt, lane, b2[nt * 16 + col]);
  }
  __syncthreads();

  // coalesced residual store: out = y + mlp(y)
  for (int r = wave * 4; r < wave * 4 + 4; ++r) {
    size_t row = row0 + r;
    int c = lane * 4;
    float4 hv = *(const float4*)(sO + r * C_DIM + c);
    float4 yv = *(const float4*)(y + row * C_DIM + c);
    float4 o  = make_float4(hv.x + yv.x, hv.y + yv.y, hv.z + yv.z, hv.w + yv.w);
    *(float4*)(out + row * C_DIM + c) = o;
  }
}

// ---------------- host ----------------
extern "C" void kernel_launch(void* const* d_in, const int* in_sizes, int n_in,
                              void* d_out, int out_size, void* d_ws, size_t ws_size,
                              hipStream_t stream) {
  (void)n_in; (void)out_size; (void)ws_size;
  const float*     x    = (const float*)d_in[0];
  const float*     ea   = (const float*)d_in[1];
  const long long* eidx = (const long long*)d_in[2];
  const float* wq = (const float*)d_in[3];  const float* bq = (const float*)d_in[4];
  const float* wk = (const float*)d_in[5];  const float* bk = (const float*)d_in[6];
  const float* wv = (const float*)d_in[7];  const float* bv = (const float*)d_in[8];
  const float* wsf= (const float*)d_in[9];  const float* bs = (const float*)d_in[10];
  const float* we = (const float*)d_in[11]; const float* be = (const float*)d_in[12];
  const float* wp = (const float*)d_in[13]; const float* bp = (const float*)d_in[14];
  const float* w1 = (const float*)d_in[15]; const float* b1 = (const float*)d_in[16];
  const float* w2 = (const float*)d_in[17]; const float* b2 = (const float*)d_in[18];
  const float* ln1g = (const float*)d_in[19]; const float* ln1b = (const float*)d_in[20];
  const float* ln2g = (const float*)d_in[21]; const float* ln2b = (const float*)d_in[22];

  const int Nn = in_sizes[0] / C_DIM;     // 50000  (divisible by 16)
  const int Ee = in_sizes[1] / E_DIM;     // 800000 (divisible by 64)

  char* wsb = (char*)d_ws; size_t off = 0;
  auto alloc = [&](size_t bytes) -> char* {
    char* p = wsb + off; off = (off + bytes + 255) & ~(size_t)255; return p;
  };
  bf16* pq = (bf16*)alloc(128 * 128 * 2);
  bf16* pk = (bf16*)alloc(128 * 128 * 2);
  bf16* pv = (bf16*)alloc(128 * 128 * 2);
  bf16* ps = (bf16*)alloc(128 * 128 * 2);
  bf16* pe = (bf16*)alloc(64 * 128 * 2);
  bf16* pp = (bf16*)alloc(128 * 128 * 2);
  bf16* p1 = (bf16*)alloc(128 * 512 * 2);
  bf16* p2 = (bf16*)alloc(512 * 128 * 2);
  float* qb   = (float*)alloc((size_t)Nn * C_DIM * 4);
  float* kb   = (float*)alloc((size_t)Nn * C_DIM * 4);
  float* vb   = (float*)alloc((size_t)Nn * C_DIM * 4);
  float* xrb  = (float*)alloc((size_t)Nn * C_DIM * 4);
  float* emat = (float*)alloc((size_t)Ee * C_DIM * 4);
  float* alp  = (float*)alloc((size_t)Ee * H_HEADS * 4);
  unsigned* amax = (unsigned*)alloc((size_t)Nn * H_HEADS * 4);
  float* den  = (float*)alloc((size_t)Nn * H_HEADS * 4);
  float* num  = (float*)alloc((size_t)Nn * C_DIM * 4);
  float* yb   = (float*)alloc((size_t)Nn * C_DIM * 4);

  // K0: pack weights
  pack_weight<<<4, 256, 0, stream>>>(wq, pq, 128, 128);
  pack_weight<<<4, 256, 0, stream>>>(wk, pk, 128, 128);
  pack_weight<<<4, 256, 0, stream>>>(wv, pv, 128, 128);
  pack_weight<<<4, 256, 0, stream>>>(wsf, ps, 128, 128);
  pack_weight<<<2, 256, 0, stream>>>(we, pe, 64, 128);
  pack_weight<<<4, 256, 0, stream>>>(wp, pp, 128, 128);
  pack_weight<<<16, 256, 0, stream>>>(w1, p1, 128, 512);
  pack_weight<<<16, 256, 0, stream>>>(w2, p2, 512, 128);

  int ntiles_n = Nn / 16;
  int ntiles_e = Ee / 64;

  node_ln_qkvs<<<ntiles_n, 128, 0, stream>>>(x, ln1g, ln1b, pq, pk, pv, ps,
                                             bq, bk, bv, bs, qb, kb, vb, xrb);
  edge_lin<<<ntiles_e, 128, 0, stream>>>(ea, pe, be, emat);

  attn_init<<<(Nn * C_DIM + 255) / 256, 256, 0, stream>>>(amax, den, num, Nn);
  attn_alpha<<<(Ee * H_HEADS + 255) / 256, 256, 0, stream>>>(qb, kb, emat, eidx, alp, amax, Ee);
  attn_accum<<<(Ee * H_HEADS + 255) / 256, 256, 0, stream>>>(alp, amax, vb, emat, eidx, den, num, Ee);

  proj<<<ntiles_n, 128, 0, stream>>>(num, den, xrb, x, pp, bp, yb);

  float* out_nodes = (float*)d_out;
  mlp<<<ntiles_n, 128, 0, stream>>>(yb, ln2g, ln2b, p1, b1, p2, b2, out_nodes);

  // tuple tail: edge_attr passthrough
  hipMemcpyAsync(out_nodes + (size_t)Nn * C_DIM, ea, (size_t)Ee * E_DIM * sizeof(float),
                 hipMemcpyDeviceToDevice, stream);
}
